// RXN_CMPD_Encoder_77043123356002
// MI455X (gfx1250) — matrix operand order repaired and checked by reading, not compiled, not run on hardware
//
#include <hip/hip_runtime.h>
#include <hip/hip_bf16.h>

#define HIDDEN    300
#define ATOM_FDIM 133
#define BOND_FDIM 147
#define MAX_NB    6
#define N_ATOMS   100000
#define N_BONDS   200000
#define N_MOLS    5000
#define DEPTH     3
#define H4        (HIDDEN / 4)      // 75 float4s per 300-wide f32 row

#define MB        (N_BONDS + 1)     // 200001
#define MA        (N_ATOMS + 1)     // 100001
#define MPADB     200064            // ceil(MB/128)*128
#define MPADA     100096            // ceil(MA/128)*128
#define KP_I      160               // padded K for W_i GEMM (147)
#define KP_H      320               // padded K for W_h GEMM (300)
#define KP_O      448               // padded K for W_o GEMM (433)

typedef __attribute__((ext_vector_type(16))) __bf16 v16bf;
typedef __attribute__((ext_vector_type(8)))  float  v8f;

// round-to-nearest-even f32 -> bf16 bits
__device__ __forceinline__ unsigned short f2bf(float f) {
  union { float f; unsigned u; } x; x.f = f;
  unsigned r = x.u + 0x7FFFu + ((x.u >> 16) & 1u);
  return (unsigned short)(r >> 16);
}
__device__ __forceinline__ unsigned pack2(float a, float b) {
  return (unsigned)f2bf(a) | ((unsigned)f2bf(b) << 16);
}

// ---------------------------------------------------------------------------
// WMMA GEMM: C[M x N] = relu( Abf(bf16, Mpad x KPAD) @ Bstage + addmat? + bias? )
// Block: 8 waves = 128 rows x 64 cols. Wave: 16 rows x 4 N-tiles.
// B (f32, tiny) staged once per block into LDS in B-fragment order
// (col-major, K-contiguous bf16). perm!=0 applies the W_o concat row remap:
//   staged_row(k) = k<300 ? k+133 : (k<433 ? k-300 : zero)
// Inner loop is branch-free: 2 global b128 (A frag) + per tile 2 ds b128 + wmma.
// ---------------------------------------------------------------------------
template <int KPAD>
__global__ __launch_bounds__(256)
void wmma_gemm_bf16(const unsigned short* __restrict__ Abf,
                    const float* __restrict__ B, int ldb, int Kact, int perm,
                    const float* __restrict__ addmat, int ldadd,
                    const float* __restrict__ bias,
                    float* __restrict__ C, int ldc, int M, int N) {
  constexpr int BS = KPAD + 8;          // LDS column stride (mult of 8 -> 16B aligned)
  __shared__ unsigned short ldsB[64 * BS];

  const int gcol0 = blockIdx.x * 64;

  // ---- stage B panel (KPAD x 64 cols) into LDS ----
  for (int flat = threadIdx.x; flat < 64 * KPAD; flat += 256) {
    int col = flat / KPAD;
    int k   = flat - col * KPAD;
    int gc  = gcol0 + col;
    int bk  = k;
    bool valid = (gc < N);
    if (perm) {
      if (k < HIDDEN)            bk = k + ATOM_FDIM;
      else if (k < Kact)         bk = k - HIDDEN;
      else                       valid = false;
    } else {
      valid = valid && (k < Kact);
    }
    float v = valid ? B[(size_t)bk * ldb + gc] : 0.f;
    ldsB[col * BS + k] = f2bf(v);
  }
  __syncthreads();

  const int wave  = threadIdx.x >> 5;
  const int lane  = threadIdx.x & 31;
  const int mbase = blockIdx.y * 128 + wave * 16;
  const int row   = mbase + (lane & 15);        // < Mpad by construction (pads zeroed)
  const int khalf = (lane >> 4) << 3;           // A: 0 or 8
  const int kbB   = (lane >> 4) << 3;           // B: 0 or 8
  const int colB  = lane & 15;

  v8f acc[4] = {};
  const unsigned short* arowp = Abf + (size_t)row * KPAD;

  for (int kc = 0; kc < KPAD; kc += 32) {
    union { uint4 u[2]; v16bf v; } af;
    af.u[0] = *reinterpret_cast<const uint4*>(arowp + kc + khalf);        // k: kc+kh..+7
    af.u[1] = *reinterpret_cast<const uint4*>(arowp + kc + 16 + khalf);   // k: kc+16+kh..+7
    #pragma unroll
    for (int t = 0; t < 4; ++t) {
      union { uint4 u[2]; v16bf v; } bf;
      const unsigned short* bsrc = &ldsB[(t * 16 + colB) * BS + kc + kbB];
      bf.u[0] = *reinterpret_cast<const uint4*>(bsrc);
      bf.u[1] = *reinterpret_cast<const uint4*>(bsrc + 16);
      acc[t] = __builtin_amdgcn_wmma_f32_16x16x32_bf16(false, af.v, false, bf.v,
                                                       (short)0, acc[t], false, false);
    }
  }

  // ---- epilogue (C layout: VGPR i -> row mbase + i + (lane>=16?8:0), col = lane&15) ----
  const int rbase = mbase + ((lane >> 4) << 3);
  #pragma unroll
  for (int t = 0; t < 4; ++t) {
    int col = gcol0 + t * 16 + (lane & 15);
    if (col < N) {
      #pragma unroll
      for (int i = 0; i < 8; ++i) {
        int r = rbase + i;
        if (r < M) {
          float v = acc[t][i];
          if (addmat) v += addmat[(size_t)r * ldadd + col];
          if (bias)   v += bias[col];
          C[(size_t)r * ldc + col] = fmaxf(v, 0.f);
        }
      }
    }
  }
}

// ---------------------------------------------------------------------------
// One-shot converters (per launch)
// ---------------------------------------------------------------------------
// f_bonds (MB x 147 f32) -> fbB (MPADB x 160 bf16), zero-padded; b128 stores.
__global__ __launch_bounds__(256)
void conv_bonds_kernel(const float* __restrict__ fb, unsigned short* __restrict__ dst) {
  long idx = (long)blockIdx.x * blockDim.x + threadIdx.x;   // MPADB * 20
  if (idx >= (long)MPADB * (KP_I / 8)) return;
  int row = (int)(idx / (KP_I / 8));
  int j8  = ((int)(idx % (KP_I / 8))) * 8;
  float v[8];
  #pragma unroll
  for (int j = 0; j < 8; ++j) {
    int k = j8 + j;
    v[j] = (row < MB && k < BOND_FDIM) ? fb[(size_t)row * BOND_FDIM + k] : 0.f;
  }
  uint4 o;
  o.x = pack2(v[0], v[1]); o.y = pack2(v[2], v[3]);
  o.z = pack2(v[4], v[5]); o.w = pack2(v[6], v[7]);
  *reinterpret_cast<uint4*>(dst + (size_t)row * KP_I + j8) = o;
}

// f_atoms (MA x 133 f32) -> aio cols [300..432] (bf16, stride 448). Pads pre-zeroed.
__global__ __launch_bounds__(256)
void conv_atoms_kernel(const float* __restrict__ fa, unsigned short* __restrict__ aio) {
  long idx = (long)blockIdx.x * blockDim.x + threadIdx.x;   // MA * 133
  if (idx >= (long)MA * ATOM_FDIM) return;
  int a = (int)(idx / ATOM_FDIM);
  int c = (int)(idx % ATOM_FDIM);
  aio[(size_t)a * KP_O + HIDDEN + c] = f2bf(fa[(size_t)a * ATOM_FDIM + c]);
}

// ---------------------------------------------------------------------------
// Message-passing gathers
// ---------------------------------------------------------------------------
// a_message[a] = sum_j message[a2b[a][j]]   (f32 out)
__global__ __launch_bounds__(256)
void atom_sum_kernel(const float* __restrict__ msg, const int* __restrict__ a2b,
                     float* __restrict__ amsg) {
  long idx = (long)blockIdx.x * blockDim.x + threadIdx.x;
  if (idx >= (long)MA * H4) return;
  int a = (int)(idx / H4), h4 = (int)(idx % H4);
  float4 s = make_float4(0.f, 0.f, 0.f, 0.f);
  #pragma unroll
  for (int j = 0; j < MAX_NB; ++j) {
    int b = a2b[(size_t)a * MAX_NB + j];
    float4 v = reinterpret_cast<const float4*>(msg + (size_t)b * HIDDEN)[h4];
    s.x += v.x; s.y += v.y; s.z += v.z; s.w += v.w;
  }
  reinterpret_cast<float4*>(amsg + (size_t)a * HIDDEN)[h4] = s;
}

// same gather-sum, but emit bf16 into aio cols [0..299] (stride 448) for the W_o GEMM
__global__ __launch_bounds__(256)
void atom_sum_bf16_kernel(const float* __restrict__ msg, const int* __restrict__ a2b,
                          unsigned short* __restrict__ aio) {
  long idx = (long)blockIdx.x * blockDim.x + threadIdx.x;
  if (idx >= (long)MA * H4) return;
  int a = (int)(idx / H4), h4 = (int)(idx % H4);
  float4 s = make_float4(0.f, 0.f, 0.f, 0.f);
  #pragma unroll
  for (int j = 0; j < MAX_NB; ++j) {
    int b = a2b[(size_t)a * MAX_NB + j];
    float4 v = reinterpret_cast<const float4*>(msg + (size_t)b * HIDDEN)[h4];
    s.x += v.x; s.y += v.y; s.z += v.z; s.w += v.w;
  }
  uint2 o; o.x = pack2(s.x, s.y); o.y = pack2(s.z, s.w);
  *reinterpret_cast<uint2*>(aio + (size_t)a * KP_O + h4 * 4) = o;
}

// tmpB[b] = bf16( a_message[b2a[b]] - message[b2revb[b]] )  (stride 320, pads pre-zeroed)
__global__ __launch_bounds__(256)
void bond_update_kernel(const float* __restrict__ amsg, const float* __restrict__ msg,
                        const int* __restrict__ b2a, const int* __restrict__ b2revb,
                        unsigned short* __restrict__ tmpB) {
  long idx = (long)blockIdx.x * blockDim.x + threadIdx.x;
  if (idx >= (long)MB * H4) return;
  int b = (int)(idx / H4), h4 = (int)(idx % H4);
  int a = b2a[b], rb = b2revb[b];
  float4 u = reinterpret_cast<const float4*>(amsg + (size_t)a  * HIDDEN)[h4];
  float4 w = reinterpret_cast<const float4*>(msg  + (size_t)rb * HIDDEN)[h4];
  uint2 o; o.x = pack2(u.x - w.x, u.y - w.y); o.y = pack2(u.z - w.z, u.w - w.w);
  *reinterpret_cast<uint2*>(tmpB + (size_t)b * KP_H + h4 * 4) = o;
}

// Deterministic segment mean over sorted mol_ids (binary search, fixed-order sum).
__global__ __launch_bounds__(256)
void mol_mean_kernel(const float* __restrict__ hid, const int* __restrict__ mol_ids,
                     float* __restrict__ out) {
  long idx = (long)blockIdx.x * blockDim.x + threadIdx.x;
  if (idx >= (long)N_MOLS * H4) return;
  int m = (int)(idx / H4), h4 = (int)(idx % H4);
  int lo = 0, hi = N_ATOMS;
  while (lo < hi) { int mid = (lo + hi) >> 1; if (mol_ids[mid] <  m) lo = mid + 1; else hi = mid; }
  int start = lo;
  hi = N_ATOMS;
  while (lo < hi) { int mid = (lo + hi) >> 1; if (mol_ids[mid] <= m) lo = mid + 1; else hi = mid; }
  int end = lo;
  float4 s = make_float4(0.f, 0.f, 0.f, 0.f);
  for (int i = start; i < end; ++i) {
    float4 v = reinterpret_cast<const float4*>(hid + (size_t)(i + 1) * HIDDEN)[h4];
    s.x += v.x; s.y += v.y; s.z += v.z; s.w += v.w;
  }
  float cnt = fmaxf((float)(end - start), 1.0f);
  reinterpret_cast<float4*>(out + (size_t)m * HIDDEN)[h4] =
      make_float4(s.x / cnt, s.y / cnt, s.z / cnt, s.w / cnt);
}

static inline size_t align256(size_t x) { return (x + 255) & ~(size_t)255; }

extern "C" void kernel_launch(void* const* d_in, const int* in_sizes, int n_in,
                              void* d_out, int out_size, void* d_ws, size_t ws_size,
                              hipStream_t stream) {
  const float* f_atoms = (const float*)d_in[0];
  const float* f_bonds = (const float*)d_in[1];
  const float* W_i     = (const float*)d_in[2];
  const float* W_h     = (const float*)d_in[3];
  const float* W_o     = (const float*)d_in[4];
  const float* b_o     = (const float*)d_in[5];
  const int*   a2b     = (const int*)d_in[6];
  const int*   b2a     = (const int*)d_in[7];
  const int*   b2revb  = (const int*)d_in[8];
  const int*   mol_ids = (const int*)d_in[9];
  float*       out     = (float*)d_out;
  (void)in_sizes; (void)n_in; (void)out_size; (void)ws_size;

  // ---- workspace layout ----
  char* ws = (char*)d_ws;
  size_t szInp  = (size_t)MB * HIDDEN * sizeof(float);            // 240 MB
  size_t szMsg  = szInp;                                          // 240 MB
  size_t szAmsg = (size_t)MA * HIDDEN * sizeof(float);            // 120 MB
  size_t szFbB  = (size_t)MPADB * KP_I * sizeof(unsigned short);  //  64 MB
  size_t szTmpB = (size_t)MPADB * KP_H * sizeof(unsigned short);  // 128 MB
  size_t szAio  = (size_t)MPADA * KP_O * sizeof(unsigned short);  //  90 MB

  size_t off = 0;
  float* inp           = (float*)(ws + off);          off += align256(szInp);
  float* msg           = (float*)(ws + off);          off += align256(szMsg);
  float* amsg          = (float*)(ws + off);          off += align256(szAmsg);
  unsigned short* fbB  = (unsigned short*)(ws + off); off += align256(szFbB);
  unsigned short* tmpB = (unsigned short*)(ws + off); off += align256(szTmpB);
  unsigned short* aio  = (unsigned short*)(ws + off); off += align256(szAio);
  float* hid = (float*)tmpB;   // tmpB dead before the W_o GEMM; 128MB >= 120MB

  dim3 blk(256);
  dim3 gB(5, MPADB / 128);   // ceil(300/64)=5 col-blocks
  dim3 gA(5, MPADA / 128);
  int nAS = (int)(((long)MA * H4 + 255) / 256);
  int nBU = (int)(((long)MB * H4 + 255) / 256);

  // zero bf16 buffers once: establishes zero K/row padding (kernels only write data region)
  hipMemsetAsync(tmpB, 0, szTmpB, stream);
  hipMemsetAsync(aio,  0, szAio,  stream);

  // pre-convert constant inputs to padded bf16
  int nCB = (int)(((long)MPADB * (KP_I / 8) + 255) / 256);
  conv_bonds_kernel<<<nCB, blk, 0, stream>>>(f_bonds, fbB);
  int nCA = (int)(((long)MA * ATOM_FDIM + 255) / 256);
  conv_atoms_kernel<<<nCA, blk, 0, stream>>>(f_atoms, aio);

  // inp = relu(f_bonds @ W_i)
  wmma_gemm_bf16<KP_I><<<gB, blk, 0, stream>>>(fbB, W_i, HIDDEN, BOND_FDIM, 0,
                                               nullptr, 0, nullptr,
                                               inp, HIDDEN, MB, HIDDEN);

  const float* cur = inp;
  for (int d = 0; d < DEPTH - 1; ++d) {
    atom_sum_kernel<<<nAS, blk, 0, stream>>>(cur, a2b, amsg);
    bond_update_kernel<<<nBU, blk, 0, stream>>>(amsg, cur, b2a, b2revb, tmpB);
    // msg = relu(inp + tmp @ W_h)
    wmma_gemm_bf16<KP_H><<<gB, blk, 0, stream>>>(tmpB, W_h, HIDDEN, HIDDEN, 0,
                                                 inp, HIDDEN, nullptr,
                                                 msg, HIDDEN, MB, HIDDEN);
    cur = msg;
  }

  // final atom aggregation straight into the bf16 concat buffer (cols 0..299)
  atom_sum_bf16_kernel<<<nAS, blk, 0, stream>>>(cur, a2b, aio);

  // hid = relu([amsg | f_atoms](perm) @ W_o + b_o)
  wmma_gemm_bf16<KP_O><<<gA, blk, 0, stream>>>(aio, W_o, HIDDEN, ATOM_FDIM + HIDDEN, 1,
                                               nullptr, 0, b_o,
                                               hid, HIDDEN, MA, HIDDEN);

  int nP = (int)(((long)N_MOLS * H4 + 255) / 256);
  mol_mean_kernel<<<nP, blk, 0, stream>>>(hid, mol_ids, out);
}
